// CPM_PARQ_47906065219889
// MI455X (gfx1250) — compile-verified, matrix operand order checked by reading, not run on hardware
//
#include <hip/hip_runtime.h>
#include <hip/hip_bf16.h>
#include <math.h>

#define BB 8
#define NBOXN 16
#define FD 32
#define FH 64
#define FW 64
#define ATOT 131072            // 32*64*64 = 2^17
#define TOPKN 7
#define NSELN 189              // (26+1)*7
#define NUMNEG 10000
#define NUMHARD 100
#define RATIOK 100

typedef __attribute__((ext_vector_type(16))) _Float16 v16h;
typedef __attribute__((ext_vector_type(8)))  float    v8f;

// ---- per-batch state block (32-bit words at ws+8192) ----
#define S_PF     0     // float  P per batch (from WMMA reduce)
#define S_POSSUM 8     // float  sum of positive losses
#define S_NEGSUM 16    // float  sum of strictly-above-threshold negative losses
#define S_P1     24    // u32    stage-1 radix prefix / final threshold key
#define S_K1     32    // i32    stage-1 remaining k / final tie count
#define S_P2     40    // u32    stage-2 prefix / threshold (inverted loss bits)
#define S_K2     48    // i32    stage-2 remaining k / final tie count
#define S_REG    64    // float  global reg sum
#define S_OFFL   65    // float  global offset sum
#define S_DIOU   66    // float  global diou sum
#define S_HIST   128   // i32[8*256]
#define STATE_WORDS (S_HIST + BB*256)

// ---------------- box preprocessing ----------------
__global__ void k_pre(const float* __restrict__ ann, float* __restrict__ annNew,
                      float* __restrict__ paintB){
  int t = threadIdx.x;
  if (t >= BB*NBOXN) return;
  const float* a = ann + t*7;
  float c0=a[0], c1=a[1], c2=a[2], d0=a[3], d1=a[4], d2=a[5];
  bool valid = a[6] > -1.0f;
  float lo0=fmaxf(c0-0.5f*d0,0.f), lo1=fmaxf(c1-0.5f*d1,0.f), lo2=fmaxf(c2-0.5f*d2,0.f);
  float hi0=fminf(c0+0.5f*d0,128.f), hi1=fminf(c1+0.5f*d1,256.f), hi2=fminf(c2+0.5f*d2,256.f);
  float n0=fmaxf(hi0-lo0,0.f), n1=fmaxf(hi1-lo1,0.f), n2=fmaxf(hi2-lo2,0.f);
  float vol = n0*n1*n2;
  float pct = vol/(d0*d1*d2);
  bool keep = valid && (vol>0.f) && (pct>0.1f) && (vol>=15.f);
  float* o = annNew + t*8;
  if (keep){ o[0]=lo0+0.5f*n0; o[1]=lo1+0.5f*n1; o[2]=lo2+0.5f*n2;
             o[3]=n0; o[4]=n1; o[5]=n2; o[6]=1.f; }
  else { o[0]=-1.f;o[1]=-1.f;o[2]=-1.f;o[3]=-1.f;o[4]=-1.f;o[5]=-1.f; o[6]=0.f; }
  o[7]=0.f;
  bool paint = valid && (vol>0.f) && !keep;
  float* p = paintB + t*8;
  p[0]=floorf(lo0); p[1]=floorf(lo1); p[2]=floorf(lo2);
  p[3]=ceilf(hi0);  p[4]=ceilf(hi1);  p[5]=ceilf(hi2);
  p[6]=paint?1.f:0.f; p[7]=0.f;
}

__global__ void k_zero(int* statei){
  for(int i=threadIdx.x;i<STATE_WORDS;i+=blockDim.x) statei[i]=0;
  __syncthreads();
  if(threadIdx.x<BB) statei[S_K1+threadIdx.x]=NUMNEG;
}

// per-anchor init + painted mask (reference compares FEAT index vs crop-coord bounds)
__global__ void k_anchor_init(const float* __restrict__ paintB, int* posA, int* gtA, int* ignA){
  __shared__ float sp[NBOXN*8];
  int i = blockIdx.x*blockDim.x + threadIdx.x;
  int b = i >> 17;
  int ai = i & (ATOT-1);
  if(threadIdx.x < NBOXN*8) sp[threadIdx.x] = paintB[b*NBOXN*8 + threadIdx.x];
  __syncthreads();
  float z = (float)(ai>>12), y=(float)((ai>>6)&63), x=(float)(ai&63);
  int painted = 0;
  #pragma unroll
  for(int j=0;j<NBOXN;j++){
    const float* p = sp + j*8;
    if(p[6]>0.5f && z>=p[0] && z<p[3] && y>=p[1] && y<p[4] && x>=p[2] && x<p[5]) painted = 2;
  }
  ignA[i]=painted; posA[i]=0; gtA[i]=0x7fffffff;
}

// exact windowed top-189 per (batch,box): 16^3 window provably holds global top-189
__global__ void k_topk(const float* __restrict__ annNew, int* posA, int* gtA, int* ignA){
  __shared__ unsigned long long key[4096];
  int box = blockIdx.x;
  int b = box >> 4;
  int bx = box & 15;
  const float* an = annNew + box*8;
  if(an[6] < 0.5f) return;                       // uniform over block
  float cz=an[0]*0.25f, cy=an[1]*0.25f, cx=an[2]*0.25f;   // strides = 4
  int l0 = min(max((int)floorf(cz)-7, 0), FD-16);
  int l1 = min(max((int)floorf(cy)-7, 0), FH-16);
  int l2 = min(max((int)floorf(cx)-7, 0), FW-16);
  for(int t=threadIdx.x;t<4096;t+=256){
    int z=l0+(t>>8), y=l1+((t>>4)&15), x=l2+(t&15);
    float dz=cz-(float)z, dy=cy-(float)y, dx=cx-(float)x;
    float r2 = fmaf(dz,dz,fmaf(dy,dy,dx*dx));
    unsigned idx = (unsigned)((z<<12)|(y<<6)|x);
    key[t] = ((unsigned long long)__float_as_uint(r2)<<32) | idx;   // dist asc, idx asc
  }
  __syncthreads();
  for(int k=2;k<=4096;k<<=1){
    for(int j=k>>1;j>0;j>>=1){
      for(int i=threadIdx.x;i<4096;i+=256){
        int ixj=i^j;
        if(ixj>i){
          unsigned long long va=key[i], vb=key[ixj];
          bool up = ((i&k)==0);
          if((va>vb)==up){ key[i]=vb; key[ixj]=va; }
        }
      }
      __syncthreads();
    }
  }
  if(threadIdx.x<NSELN){
    int idx = (int)(key[threadIdx.x] & 0xFFFFFFFFull);
    int gi = b*ATOT + idx;
    if(threadIdx.x<TOPKN){ atomicOr(&posA[gi],1); atomicMin(&gtA[gi], bx); }
    else atomicOr(&ignA[gi],1);
  }
}

// P[b] = sum of 0/1 pos flags via ones-matmul on the matrix pipe (exact for 0/1 in f16)
__global__ void k_preduce(const int* __restrict__ posA, float* statef){
  int b = blockIdx.x;
  int wave = threadIdx.x >> 5;
  int lane = threadIdx.x & 31;
  const int* base = posA + b*ATOT + wave*(ATOT/8);
  v16h ones = {(_Float16)1.f,(_Float16)1.f,(_Float16)1.f,(_Float16)1.f,
               (_Float16)1.f,(_Float16)1.f,(_Float16)1.f,(_Float16)1.f,
               (_Float16)1.f,(_Float16)1.f,(_Float16)1.f,(_Float16)1.f,
               (_Float16)1.f,(_Float16)1.f,(_Float16)1.f,(_Float16)1.f};
  v8f acc = {0.f,0.f,0.f,0.f,0.f,0.f,0.f,0.f};
  #pragma unroll 4
  for(int it=0; it<32; ++it){                 // 512 flags per WMMA, 16384 per wave
    const int* p = base + it*512 + lane*16;
    int4 w0 = ((const int4*)p)[0];
    int4 w1 = ((const int4*)p)[1];
    int4 w2 = ((const int4*)p)[2];
    int4 w3 = ((const int4*)p)[3];
    v16h bh;
    bh[0]=(_Float16)(float)w0.x;  bh[1]=(_Float16)(float)w0.y;
    bh[2]=(_Float16)(float)w0.z;  bh[3]=(_Float16)(float)w0.w;
    bh[4]=(_Float16)(float)w1.x;  bh[5]=(_Float16)(float)w1.y;
    bh[6]=(_Float16)(float)w1.z;  bh[7]=(_Float16)(float)w1.w;
    bh[8]=(_Float16)(float)w2.x;  bh[9]=(_Float16)(float)w2.y;
    bh[10]=(_Float16)(float)w2.z; bh[11]=(_Float16)(float)w2.w;
    bh[12]=(_Float16)(float)w3.x; bh[13]=(_Float16)(float)w3.y;
    bh[14]=(_Float16)(float)w3.z; bh[15]=(_Float16)(float)w3.w;
    acc = __builtin_amdgcn_wmma_f32_16x16x32_f16(false, ones, false, bh,
                                                 (short)0, acc, false, false);
  }
  // sum of all D entries = 16 * chunk total (layout-agnostic)
  float s = acc[0]+acc[1]+acc[2]+acc[3]+acc[4]+acc[5]+acc[6]+acc[7];
  for(int o=16;o>0;o>>=1) s += __shfl_xor(s,o);
  if(lane==0) atomicAdd(&statef[S_PF+b], s*(1.0f/16.0f));
}

// focal loss per anchor (needs P for the false-negative x4 term)
__global__ void k_losselem(const float* __restrict__ Cls, const int* __restrict__ posA,
                           const int* __restrict__ ignA, float* statef,
                           float* __restrict__ lossA){
  int i = blockIdx.x*blockDim.x + threadIdx.x;
  int b = i >> 17;
  float pred = Cls[i];
  int pv = posA[i];
  float target = pv ? 1.f : 0.f;
  float prob = 1.f/(1.f+expf(-pred));
  prob = fminf(fmaxf(prob, 1e-4f), 1.f-1e-4f);
  float alpha = pv ? 0.75f : 0.25f;
  float pt = pv ? (1.f-prob) : prob;
  float focal = alpha*pt*pt;
  float bce = fmaxf(pred,0.f) - pred*target + log1pf(expf(-fabsf(pred)));
  float loss = (ignA[i]!=0) ? 0.f : focal*bce;
  if(pv && prob<0.8f && statef[S_PF+b]>0.5f) loss *= 4.f;
  lossA[i]=loss;
  if(pv) atomicAdd(&statef[S_POSSUM+b], loss);
}

// radix-threshold histogram: mode 0 = neg_rand keys, mode 1 = inverted loss keys (sampled only)
__global__ void k_hist(int mode, int level, const int* __restrict__ posA,
                       const float* __restrict__ Neg, const float* __restrict__ lossA,
                       const unsigned char* __restrict__ sampA,
                       const unsigned* __restrict__ stateu, int* statei){
  int i = blockIdx.x*blockDim.x + threadIdx.x;
  int b = i >> 17;
  unsigned key; bool act;
  if(mode==0){ act = true; key = posA[i] ? 0xFFFFFFFFu : __float_as_uint(Neg[i]); }
  else       { act = sampA[i]!=0; key = ~__float_as_uint(lossA[i]); }
  unsigned pref = stateu[(mode ? S_P2 : S_P1) + b];
  if(act && (level==0 || (key >> (8*(4-level))) == pref)){
    int bin = (int)((key >> (8*(3-level))) & 0xFFu);
    atomicAdd(&statei[S_HIST + b*256 + bin], 1);
  }
}

__global__ void k_sel(int pslot, int kslot, unsigned* stateu, int* statei){
  int b = blockIdx.x;
  if(threadIdx.x==0){
    int k = statei[kslot+b];
    unsigned pref = stateu[pslot+b];
    int cum=0, bin=255, before=0; bool found=false;
    for(int c=0;c<256;c++){
      int h = statei[S_HIST + b*256 + c];
      if(!found && cum+h >= k){ bin=c; before=cum; found=true; }
      cum += h;
    }
    if(!found){ bin=255; before = cum - statei[S_HIST + b*256 + 255]; }
    stateu[pslot+b] = (pref<<8) | (unsigned)bin;
    statei[kslot+b] = k - before;
  }
  __syncthreads();
  statei[S_HIST + b*256 + threadIdx.x] = 0;   // ready for next level/stage
}

__global__ void k_mark(const int* __restrict__ posA, const float* __restrict__ Neg,
                       const unsigned* __restrict__ stateu, unsigned char* sampA){
  int i = blockIdx.x*blockDim.x + threadIdx.x;
  int b = i >> 17;
  unsigned key = posA[i] ? 0xFFFFFFFFu : __float_as_uint(Neg[i]);
  sampA[i] = (key < stateu[S_P1+b]) ? (unsigned char)1 : (unsigned char)0;
}

// deterministic index-ordered resolution of rand ties at the sampling threshold
__global__ void k_tie(const int* __restrict__ posA, const float* __restrict__ Neg,
                      const unsigned* __restrict__ stateu, const int* __restrict__ statei,
                      unsigned char* sampA){
  __shared__ int list[64];
  __shared__ int cnt;
  int b = blockIdx.x;
  if(threadIdx.x==0) cnt=0;
  __syncthreads();
  unsigned T = stateu[S_P1+b];
  for(int a=threadIdx.x; a<ATOT; a+=blockDim.x){
    int i = b*ATOT + a;
    unsigned key = posA[i] ? 0xFFFFFFFFu : __float_as_uint(Neg[i]);
    if(key==T){ int p = atomicAdd(&cnt,1); if(p<64) list[p]=a; }
  }
  __syncthreads();
  if(threadIdx.x==0){
    int n = min(cnt,64);
    for(int u=1;u<n;u++){ int v=list[u]; int w=u-1;
      while(w>=0 && list[w]>v){ list[w+1]=list[w]; w--; } list[w+1]=v; }
    int need = statei[S_K1+b];
    if(need>n) need=n;
    for(int j=0;j<need;j++) sampA[b*ATOT + list[j]] = 1;
  }
}

__global__ void k_init2(const float* statef, int* statei, unsigned* stateu){
  int t=threadIdx.x;
  if(t<BB){
    int P = (int)(statef[S_PF+t] + 0.5f);
    int k2 = (P>0) ? min(RATIOK*P, NUMNEG) : NUMHARD;
    statei[S_K2+t]=k2;
    stateu[S_P2+t]=0u;
  }
}

__global__ void k_negsum(const float* __restrict__ lossA, const unsigned char* __restrict__ sampA,
                         const unsigned* __restrict__ stateu, float* statef){
  __shared__ float ssum;
  if(threadIdx.x==0) ssum=0.f;
  __syncthreads();
  int i = blockIdx.x*blockDim.x + threadIdx.x;
  int b = (int)((blockIdx.x*blockDim.x) >> 17);
  if(sampA[i]){
    float l = lossA[i];
    unsigned ik = ~__float_as_uint(l);
    if(ik < stateu[S_P2+b]) atomicAdd(&ssum, l);   // strictly above threshold
  }
  __syncthreads();
  if(threadIdx.x==0 && ssum!=0.f) atomicAdd(&statef[S_NEGSUM+b], ssum);
}

__global__ void k_reg(const float* __restrict__ Shp, const float* __restrict__ Off,
                      const float* __restrict__ annNew, const int* __restrict__ posA,
                      const int* __restrict__ gtA, float* statef){
  __shared__ float sreg, soff, sdio;
  if(threadIdx.x==0){sreg=0.f;soff=0.f;sdio=0.f;}
  __syncthreads();
  int i = blockIdx.x*blockDim.x + threadIdx.x;
  int b = i >> 17;
  int ai = i & (ATOT-1);
  if(posA[i]){
    int g = gtA[i];
    const float* an = annNew + (b*NBOXN+g)*8;
    float c0=an[0],c1=an[1],c2=an[2],n0=an[3],n1=an[4],n2=an[5];
    float a0=(float)(ai>>12), a1=(float)((ai>>6)&63), a2=(float)(ai&63);
    float po0=Off[(b*3+0)*ATOT+ai], po1=Off[(b*3+1)*ATOT+ai], po2=Off[(b*3+2)*ATOT+ai];
    float ps0=Shp[(b*3+0)*ATOT+ai], ps1=Shp[(b*3+1)*ATOT+ai], ps2=Shp[(b*3+2)*ATOT+ai];
    float reg = fabsf(ps0-0.5f*n0)+fabsf(ps1-0.5f*n1)+fabsf(ps2-0.5f*n2);
    float to0=c0*0.25f-a0, to1=c1*0.25f-a1, to2=c2*0.25f-a2;
    float off = fabsf(po0-to0)+fabsf(po1-to1)+fabsf(po2-to2);
    float pc0=(a0+po0)*4.f, pc1=(a1+po1)*4.f, pc2=(a2+po2)*4.f;
    float pw0=2.f*ps0, pw1=2.f*ps1, pw2=2.f*ps2;
    float l10=pc0-0.5f*pw0,h10=pc0+0.5f*pw0;
    float l11=pc1-0.5f*pw1,h11=pc1+0.5f*pw1;
    float l12=pc2-0.5f*pw2,h12=pc2+0.5f*pw2;
    float l20=c0-0.5f*n0,h20=c0+0.5f*n0;
    float l21=c1-0.5f*n1,h21=c1+0.5f*n1;
    float l22=c2-0.5f*n2,h22=c2+0.5f*n2;
    float i0=fmaxf(fminf(h10,h20)-fmaxf(l10,l20),0.f);
    float i1=fmaxf(fminf(h11,h21)-fmaxf(l11,l21),0.f);
    float i2=fmaxf(fminf(h12,h22)-fmaxf(l12,l22),0.f);
    float inter=i0*i1*i2+1e-7f;
    float uni=pw0*pw1*pw2 + n0*n1*n2 - inter;
    float iou=inter/uni;
    float cd0=fmaxf(h10,h20)-fminf(l10,l20);
    float cd1=fmaxf(h11,h21)-fminf(l11,l21);
    float cd2=fmaxf(h12,h22)-fminf(l12,l22);
    float c2s=cd0*cd0+cd1*cd1+cd2*cd2+1e-7f;
    float q0=l20+h20-l10-h10, q1=l21+h21-l11-h11, q2=l22+h22-l12-h12;
    float diou = iou - 0.25f*(q0*q0+q1*q1+q2*q2)/c2s;
    atomicAdd(&sreg,reg); atomicAdd(&soff,off); atomicAdd(&sdio,diou);
  }
  __syncthreads();
  if(threadIdx.x==0){
    if(sreg!=0.f) atomicAdd(&statef[S_REG], sreg);
    if(soff!=0.f) atomicAdd(&statef[S_OFFL], soff);
    if(sdio!=0.f) atomicAdd(&statef[S_DIOU], sdio);
  }
}

__global__ void k_final(const float* statef, const int* statei, const unsigned* stateu,
                        float* out){
  if(threadIdx.x!=0) return;
  float cls=0.f, nfg=0.f;
  for(int b=0;b<BB;b++){
    float P = floorf(statef[S_PF+b]+0.5f);
    float tieval = __uint_as_float(~stateu[S_P2+b]);
    float neg = statef[S_NEGSUM+b] + tieval*(float)statei[S_K2+b];
    cls += (statef[S_POSSUM+b] + neg)/fmaxf(P,1.f);
    nfg += P;
  }
  cls *= (1.f/(float)BB);
  bool has = nfg > 0.5f;
  out[0]=cls;
  out[1]= has ? statef[S_REG]/fmaxf(3.f*nfg,1.f) : 0.f;
  out[2]= has ? statef[S_OFFL]/fmaxf(3.f*nfg,1.f) : 0.f;
  out[3]= has ? -statef[S_DIOU]/fmaxf(nfg,1.f) : 0.f;
}

extern "C" void kernel_launch(void* const* d_in, const int* in_sizes, int n_in,
                              void* d_out, int out_size, void* d_ws, size_t ws_size,
                              hipStream_t stream){
  const float* Cls = (const float*)d_in[0];
  const float* Shp = (const float*)d_in[1];
  const float* Off = (const float*)d_in[2];
  const float* Ann = (const float*)d_in[3];
  const float* Neg = (const float*)d_in[4];
  char* ws = (char*)d_ws;
  float*    annNew = (float*)(ws);                 // 4 KB
  float*    paintB = (float*)(ws + 4096);          // 4 KB
  unsigned* stateu = (unsigned*)(ws + 8192);       // <= 16 KB
  int*      statei = (int*)stateu;
  float*    statef = (float*)stateu;
  size_t off = 24576;
  int*   posA  = (int*)(ws + off);   off += (size_t)BB*ATOT*4;
  int*   gtA   = (int*)(ws + off);   off += (size_t)BB*ATOT*4;
  int*   ignA  = (int*)(ws + off);   off += (size_t)BB*ATOT*4;
  float* lossA = (float*)(ws + off); off += (size_t)BB*ATOT*4;
  unsigned char* sampA = (unsigned char*)(ws + off);
  float* outp = (float*)d_out;
  dim3 gElem(BB*ATOT/256);

  k_pre<<<1,128,0,stream>>>(Ann, annNew, paintB);
  k_zero<<<1,256,0,stream>>>(statei);
  k_anchor_init<<<gElem,256,0,stream>>>(paintB, posA, gtA, ignA);
  k_topk<<<BB*NBOXN,256,0,stream>>>(annNew, posA, gtA, ignA);
  k_preduce<<<BB,256,0,stream>>>(posA, statef);
  k_losselem<<<gElem,256,0,stream>>>(Cls, posA, ignA, statef, lossA);
  for(int lv=0; lv<4; ++lv){
    k_hist<<<gElem,256,0,stream>>>(0, lv, posA, Neg, lossA, sampA, stateu, statei);
    k_sel<<<BB,256,0,stream>>>(S_P1, S_K1, stateu, statei);
  }
  k_mark<<<gElem,256,0,stream>>>(posA, Neg, stateu, sampA);
  k_tie<<<BB,256,0,stream>>>(posA, Neg, stateu, statei, sampA);
  k_init2<<<1,32,0,stream>>>(statef, statei, stateu);
  for(int lv=0; lv<4; ++lv){
    k_hist<<<gElem,256,0,stream>>>(1, lv, posA, Neg, lossA, sampA, stateu, statei);
    k_sel<<<BB,256,0,stream>>>(S_P2, S_K2, stateu, statei);
  }
  k_negsum<<<gElem,256,0,stream>>>(lossA, sampA, stateu, statef);
  k_reg<<<gElem,256,0,stream>>>(Shp, Off, annNew, posA, gtA, statef);
  k_final<<<1,1,0,stream>>>(statef, statei, stateu, outp);
}
